// SMC_Transformer_23562190585934
// MI455X (gfx1250) — compile-verified, matrix operand order checked by reading, not run on hardware
//
#include <hip/hip_runtime.h>
#include <hip/hip_bf16.h>

// ---------------------------------------------------------------------------
// SMC transformer cell (reduces to single-head causal attention), bf16 WMMA.
// B=32, P=1, S=512, FX=8, D=256, OUT=8
// Outputs (flat, fp32): pred, pred_resampl, k, v, R, attn
// Wave tile: 64x32 (4 A-frags x 2 B-frags -> 8 WMMAs per 6 fragment loads).
// ---------------------------------------------------------------------------

typedef __attribute__((ext_vector_type(16))) __bf16 v16bf;
typedef __attribute__((ext_vector_type(8)))  __bf16 v8bf;
typedef __attribute__((ext_vector_type(8)))  float  v8f;

#define WMMA_BF16(a, b, c) \
    __builtin_amdgcn_wmma_f32_16x16x32_bf16(false, (a), false, (b), (short)0, (c), false, false)

constexpr int B_   = 32;
constexpr int S_   = 512;
constexpr int D_   = 256;
constexpr int FX_  = 8;
constexpr int OUT_ = 8;
constexpr int ROWS = B_ * S_;   // 16384

// Load a 16-element bf16 fragment per CDNA5 16-bit A/B layout:
// elements 0..7 = p[0..7], elements 8..15 = p[16..23] (two 16B loads).
__device__ inline v16bf load_frag(const __bf16* p) {
    v8bf lo = *(const v8bf*)(p);
    v8bf hi = *(const v8bf*)(p + 16);
    v16bf r;
#pragma unroll
    for (int i = 0; i < 8; ++i) { r[i] = lo[i]; r[i + 8] = hi[i]; }
    return r;
}

// ---------------------------------------------------------------------------
// Transpose + quantize the four DxD weights to bf16 (Wt[n][k] = W[k][n]).
// ---------------------------------------------------------------------------
__global__ void prep_w_kernel(const float* __restrict__ Wq, const float* __restrict__ Wk,
                              const float* __restrict__ Wv, const float* __restrict__ Wo,
                              __bf16* __restrict__ WqT, __bf16* __restrict__ WkT,
                              __bf16* __restrict__ WvT, __bf16* __restrict__ WoT) {
    int idx = blockIdx.x * 256 + threadIdx.x;          // < 4*65536
    int w = idx >> 16, e = idx & 65535;
    int n = e >> 8, k = e & 255;
    const float* s = (w == 0) ? Wq : (w == 1) ? Wk : (w == 2) ? Wv : Wo;
    __bf16*      d = (w == 0) ? WqT : (w == 1) ? WkT : (w == 2) ? WvT : WoT;
    d[n * D_ + k] = (__bf16)s[k * D_ + n];
}

// ---------------------------------------------------------------------------
// x = (inputs @ Wp + bp) * sqrt(D), quantized to bf16.  K=8 -> scalar FMA.
// ---------------------------------------------------------------------------
__global__ void proj_x_kernel(const float* __restrict__ in, const float* __restrict__ Wp,
                              const float* __restrict__ bp, __bf16* __restrict__ x_h) {
    int idx = blockIdx.x * 256 + threadIdx.x;          // < ROWS*D_
    int row = idx >> 8, d = idx & 255;
    const float* ip = in + (size_t)row * FX_;
    float s = bp[d];
#pragma unroll
    for (int f = 0; f < FX_; ++f) s += ip[f] * Wp[f * D_ + d];
    x_h[idx] = (__bf16)(s * 16.0f);                    // sqrt(256) = 16
}

// ---------------------------------------------------------------------------
// Generic batched WMMA GEMM: C[M,N] = A[M,K](bf16) * Bt[N,K]^T(bf16) * scale + bias
// Each wave: 64x32 tile.  Per K-step: 6 fragment loads feed 8 WMMAs.
// Optional fp32 and/or bf16 outputs.
// ---------------------------------------------------------------------------
__global__ __launch_bounds__(256)
void gemm_bf16_kernel(const __bf16* __restrict__ A, const __bf16* __restrict__ Bt,
                      const float* __restrict__ bias, float* __restrict__ Cf,
                      __bf16* __restrict__ Ch, int M, int N, int K,
                      long long batchA, long long batchB,
                      long long batchCf, long long batchCh, float scale) {
    int b = blockIdx.z;
    A  += (size_t)b * batchA;
    Bt += (size_t)b * batchB;
    int wave = threadIdx.x >> 5, lane = threadIdx.x & 31;
    int m0 = blockIdx.x * 64;
    int n0 = (blockIdx.y * 8 + wave) * 32;
    if (m0 >= M || n0 >= N) return;                    // wave-uniform
    int r = lane & 15, h = lane >> 4;

    const __bf16* ap  = A  + (size_t)(m0 + r) * K + h * 8;
    const __bf16* bp0 = Bt + (size_t)(n0 + r) * K + h * 8;
    const __bf16* bp1 = bp0 + (size_t)16 * K;

    v8f acc[8] = {};
    for (int k0 = 0; k0 < K; k0 += 32) {
        v16bf b0 = load_frag(bp0 + k0);
        v16bf b1 = load_frag(bp1 + k0);
        v16bf a0 = load_frag(ap + k0);
        v16bf a1 = load_frag(ap + (size_t)16 * K + k0);
        v16bf a2 = load_frag(ap + (size_t)32 * K + k0);
        v16bf a3 = load_frag(ap + (size_t)48 * K + k0);
        acc[0] = WMMA_BF16(a0, b0, acc[0]);
        acc[1] = WMMA_BF16(a1, b0, acc[1]);
        acc[2] = WMMA_BF16(a2, b0, acc[2]);
        acc[3] = WMMA_BF16(a3, b0, acc[3]);
        acc[4] = WMMA_BF16(a0, b1, acc[4]);
        acc[5] = WMMA_BF16(a1, b1, acc[5]);
        acc[6] = WMMA_BF16(a2, b1, acc[6]);
        acc[7] = WMMA_BF16(a3, b1, acc[7]);
    }

    float*  cf = Cf ? Cf + (size_t)b * batchCf : nullptr;
    __bf16* ch = Ch ? Ch + (size_t)b * batchCh : nullptr;
#pragma unroll
    for (int s = 0; s < 2; ++s) {
        int col = n0 + 16 * s + r;
        float bv = bias ? bias[col] : 0.0f;
#pragma unroll
        for (int t = 0; t < 4; ++t) {
            int rowb = m0 + 16 * t + 8 * h;
#pragma unroll
            for (int i = 0; i < 8; ++i) {
                float val = acc[4 * s + t][i] * scale + bv;
                size_t off = (size_t)(rowb + i) * N + col;
                if (cf) cf[off] = val;
                if (ch) ch[off] = (__bf16)val;
            }
        }
    }
}

// ---------------------------------------------------------------------------
// scores = q @ k^T / sqrt(D) with causal mask, written fp32 into attn region.
// Row-major k is already the transposed B operand (Bt[n][kdim] = k[n][kdim]).
// Fully-masked 64x32 wave tiles skip the WMMA loop entirely.
// ---------------------------------------------------------------------------
__global__ __launch_bounds__(256)
void scores_kernel(const __bf16* __restrict__ qh, const __bf16* __restrict__ kh,
                   float* __restrict__ attn, float scale) {
    int b = blockIdx.z;
    const __bf16* A  = qh + (size_t)b * S_ * D_;
    const __bf16* Bt = kh + (size_t)b * S_ * D_;
    float*        C  = attn + (size_t)b * S_ * S_;
    int wave = threadIdx.x >> 5, lane = threadIdx.x & 31;
    int m0 = blockIdx.x * 64;
    int n0 = (blockIdx.y * 8 + wave) * 32;
    int r = lane & 15, h = lane >> 4;

    if (n0 > m0 + 63) {                                // tile entirely above diagonal
#pragma unroll
        for (int s = 0; s < 2; ++s) {
            int col = n0 + 16 * s + r;
#pragma unroll
            for (int t = 0; t < 4; ++t) {
                int rowb = m0 + 16 * t + 8 * h;
#pragma unroll
                for (int i = 0; i < 8; ++i) C[(size_t)(rowb + i) * S_ + col] = -1e9f;
            }
        }
        return;
    }

    const __bf16* ap  = A  + (size_t)(m0 + r) * D_ + h * 8;
    const __bf16* bp0 = Bt + (size_t)(n0 + r) * D_ + h * 8;
    const __bf16* bp1 = bp0 + (size_t)16 * D_;
    v8f acc[8] = {};
    for (int k0 = 0; k0 < D_; k0 += 32) {
        v16bf b0 = load_frag(bp0 + k0);
        v16bf b1 = load_frag(bp1 + k0);
        v16bf a0 = load_frag(ap + k0);
        v16bf a1 = load_frag(ap + (size_t)16 * D_ + k0);
        v16bf a2 = load_frag(ap + (size_t)32 * D_ + k0);
        v16bf a3 = load_frag(ap + (size_t)48 * D_ + k0);
        acc[0] = WMMA_BF16(a0, b0, acc[0]);
        acc[1] = WMMA_BF16(a1, b0, acc[1]);
        acc[2] = WMMA_BF16(a2, b0, acc[2]);
        acc[3] = WMMA_BF16(a3, b0, acc[3]);
        acc[4] = WMMA_BF16(a0, b1, acc[4]);
        acc[5] = WMMA_BF16(a1, b1, acc[5]);
        acc[6] = WMMA_BF16(a2, b1, acc[6]);
        acc[7] = WMMA_BF16(a3, b1, acc[7]);
    }
#pragma unroll
    for (int s = 0; s < 2; ++s) {
        int col = n0 + 16 * s + r;
#pragma unroll
        for (int t = 0; t < 4; ++t) {
            int rowb = m0 + 16 * t + 8 * h;
#pragma unroll
            for (int i = 0; i < 8; ++i) {
                int row = rowb + i;
                C[(size_t)row * S_ + col] =
                    (col <= row) ? acc[4 * s + t][i] * scale : -1e9f;
            }
        }
    }
}

// ---------------------------------------------------------------------------
// Row softmax: one wave32 per 512-wide row (16 elems/lane), in-place fp32
// plus bf16 copy for the z GEMM.
// ---------------------------------------------------------------------------
__global__ __launch_bounds__(256)
void softmax_kernel(float* __restrict__ attn, __bf16* __restrict__ attn_h) {
    int rowg = blockIdx.x * 8 + (threadIdx.x >> 5);    // < B_*S_
    int lane = threadIdx.x & 31;
    float* p = attn + (size_t)rowg * S_;
    float v[16];
    float mx = -3.4e38f;
#pragma unroll
    for (int i = 0; i < 16; ++i) { v[i] = p[lane * 16 + i]; mx = fmaxf(mx, v[i]); }
#pragma unroll
    for (int o = 16; o > 0; o >>= 1) mx = fmaxf(mx, __shfl_xor(mx, o, 32));
    float sum = 0.0f;
#pragma unroll
    for (int i = 0; i < 16; ++i) { v[i] = __expf(v[i] - mx); sum += v[i]; }
#pragma unroll
    for (int o = 16; o > 0; o >>= 1) sum += __shfl_xor(sum, o, 32);
    float inv = 1.0f / sum;
    __bf16* ph = attn_h + (size_t)rowg * S_;
#pragma unroll
    for (int i = 0; i < 16; ++i) {
        float z = v[i] * inv;
        p[lane * 16 + i]  = z;
        ph[lane * 16 + i] = (__bf16)z;
    }
}

// ---------------------------------------------------------------------------
// vT[b][d][s] = v[b][s][d] quantized to bf16 (B operand for z = attn @ v).
// ---------------------------------------------------------------------------
__global__ void transpose_v_kernel(const float* __restrict__ v, __bf16* __restrict__ vT) {
    int idx = blockIdx.x * 256 + threadIdx.x;          // < B_*D_*S_
    int b = idx >> 17;                                 // D_*S_ = 131072 = 2^17
    int rem = idx & 131071;
    int d = rem >> 9;                                  // S_ = 512
    int s = rem & 511;
    vT[idx] = (__bf16)v[(size_t)b * S_ * D_ + (size_t)s * D_ + d];
}

// ---------------------------------------------------------------------------
// pred = R @ Wf + bf (K=256, OUT=8); also duplicates into pred_resampl.
// ---------------------------------------------------------------------------
__global__ void pred_kernel(const float* __restrict__ R, const float* __restrict__ Wf,
                            const float* __restrict__ bf_, float* __restrict__ pred,
                            float* __restrict__ pred_r) {
    int idx = blockIdx.x * 256 + threadIdx.x;          // < ROWS*OUT_
    int row = idx >> 3, o = idx & 7;
    const float* r = R + (size_t)row * D_;
    float s = bf_[o];
    for (int d = 0; d < D_; ++d) s += r[d] * Wf[d * OUT_ + o];
    pred[idx]   = s;
    pred_r[idx] = s;
}

// ---------------------------------------------------------------------------
extern "C" void kernel_launch(void* const* d_in, const int* in_sizes, int n_in,
                              void* d_out, int out_size, void* d_ws, size_t ws_size,
                              hipStream_t stream) {
    (void)in_sizes; (void)n_in; (void)out_size; (void)ws_size;
    const float* inputs = (const float*)d_in[0];
    // d_in[1] = targets (unused)
    const float* Wp = (const float*)d_in[2];  const float* bp = (const float*)d_in[3];
    const float* Wq = (const float*)d_in[4];  const float* bq = (const float*)d_in[5];
    const float* Wk = (const float*)d_in[6];  const float* bk = (const float*)d_in[7];
    const float* Wv = (const float*)d_in[8];  const float* bv = (const float*)d_in[9];
    const float* Wo = (const float*)d_in[10]; const float* bo = (const float*)d_in[11];
    const float* Wf = (const float*)d_in[12]; const float* bf_ = (const float*)d_in[13];

    float* out     = (float*)d_out;
    float* o_pred  = out;                               // 131072
    float* o_predr = o_pred + ROWS * OUT_;              // 131072
    float* o_k     = o_predr + ROWS * OUT_;             // 4194304
    float* o_v     = o_k + (size_t)ROWS * D_;           // 4194304
    float* o_R     = o_v + (size_t)ROWS * D_;           // 4194304
    float* o_attn  = o_R + (size_t)ROWS * D_;           // 8388608

    char* w = (char*)d_ws;                              // all 16B-aligned regions
    __bf16* x_h    = (__bf16*)(w);                      //  8 MiB
    __bf16* q_h    = (__bf16*)(w + (size_t)8  * 1048576);
    __bf16* k_h    = (__bf16*)(w + (size_t)16 * 1048576);
    __bf16* vT_h   = (__bf16*)(w + (size_t)24 * 1048576);
    __bf16* attn_h = (__bf16*)(w + (size_t)32 * 1048576); // 16 MiB
    __bf16* z_h    = (__bf16*)(w + (size_t)48 * 1048576);
    __bf16* WqT    = (__bf16*)(w + (size_t)56 * 1048576);
    __bf16* WkT    = WqT + 65536;
    __bf16* WvT    = WkT + 65536;
    __bf16* WoT    = WvT + 65536;

    prep_w_kernel<<<1024, 256, 0, stream>>>(Wq, Wk, Wv, Wo, WqT, WkT, WvT, WoT);
    proj_x_kernel<<<ROWS * D_ / 256, 256, 0, stream>>>(inputs, Wp, bp, x_h);

    // Block covers 64 rows x 256 cols (8 waves x 64x32 tiles).
    dim3 g1(ROWS / 64, 1, 1);
    gemm_bf16_kernel<<<g1, 256, 0, stream>>>(x_h, WqT, bq, nullptr, q_h,
                                             ROWS, D_, D_, 0, 0, 0, 0, 1.0f);
    gemm_bf16_kernel<<<g1, 256, 0, stream>>>(x_h, WkT, bk, o_k, k_h,
                                             ROWS, D_, D_, 0, 0, 0, 0, 1.0f);
    gemm_bf16_kernel<<<g1, 256, 0, stream>>>(x_h, WvT, bv, o_v, nullptr,
                                             ROWS, D_, D_, 0, 0, 0, 0, 1.0f);
    transpose_v_kernel<<<B_ * D_ * S_ / 256, 256, 0, stream>>>(o_v, vT_h);

    dim3 gs(S_ / 64, S_ / 256, B_);
    scores_kernel<<<gs, 256, 0, stream>>>(q_h, k_h, o_attn, 1.0f / 16.0f);
    softmax_kernel<<<B_ * S_ / 8, 256, 0, stream>>>(o_attn, attn_h);

    dim3 gz(S_ / 64, 1, B_);
    gemm_bf16_kernel<<<gz, 256, 0, stream>>>(attn_h, vT_h, nullptr, nullptr, z_h,
                                             S_, D_, S_,
                                             (long long)S_ * S_, (long long)D_ * S_,
                                             0, (long long)S_ * D_, 1.0f);
    gemm_bf16_kernel<<<g1, 256, 0, stream>>>(z_h, WoT, bo, o_R, nullptr,
                                             ROWS, D_, D_, 0, 0, 0, 0, 1.0f);
    pred_kernel<<<ROWS * OUT_ / 256, 256, 0, stream>>>(o_R, Wf, bf_, o_pred, o_predr);
}
